// CompGCNLayer_12180527251908
// MI455X (gfx1250) — compile-verified, matrix operand order checked by reading, not run on hardware
//
#include <hip/hip_runtime.h>
#include <hip/hip_bf16.h>

#define NN 100000
#define NE 1600000
#define DD 64

typedef __attribute__((ext_vector_type(16))) __bf16 v16bf;
typedef __attribute__((ext_vector_type(2)))  __bf16 v2bf;
typedef __attribute__((ext_vector_type(8)))  float  v8f;

// Fragment: 8 dwords = 16 bf16. p[v] holds K-pair (k, k+1) for VGPR v.
union Frag { v2bf p[8]; v16bf v; };

// pack two f32 -> 2x bf16 via hardware convert (fptrunc -> v_cvt*bf16)
__device__ __forceinline__ v2bf pk_bf16(float lo, float hi) {
    v2bf r;
    r.x = (__bf16)lo;
    r.y = (__bf16)hi;
    return r;
}

// K index for 16-bit A/B fragment VGPR v, lane-half hh, K-frame f (ISA 7.12.2)
__device__ __forceinline__ int frag_k(int f, int v, int hh) {
    return f * 32 + ((v & 4) ? 16 : 0) + hh * 8 + (v & 3) * 2;
}

__device__ __forceinline__ v8f wmma_bf(const Frag& a, const Frag& b, v8f c) {
    return __builtin_amdgcn_wmma_f32_16x16x32_bf16(
        false, a.v, false, b.v, (short)0, c, false, false);
}

// B fragments for a 64x64 row-major f32 weight matrix -> bf16.
// Bt[t][f]: output cols [16t,16t+16), K-frame f.
__device__ __forceinline__ void load_wfrags(const float* __restrict__ W,
                                            int lane, Frag Bt[4][2]) {
    const int n0 = lane & 15, hh = lane >> 4;
#pragma unroll
    for (int t = 0; t < 4; ++t)
#pragma unroll
        for (int f = 0; f < 2; ++f)
#pragma unroll
            for (int v = 0; v < 8; ++v) {
                const int k = frag_k(f, v, hh);
                const int n = t * 16 + n0;
                Bt[t][f].p[v] = pk_bf16(W[k * DD + n], W[(k + 1) * DD + n]);
            }
}

// A fragments for 16 rows from per-lane row pointer (f32, 256B-aligned rows)
__device__ __forceinline__ void load_afrags(const float2* __restrict__ row,
                                            int hh, Frag A[2]) {
#pragma unroll
    for (int f = 0; f < 2; ++f)
#pragma unroll
        for (int v = 0; v < 8; ++v) {
            const int k = frag_k(f, v, hh);
            const float2 x = row[k >> 1];
            A[f].p[v] = pk_bf16(x.x, x.y);
        }
}

__global__ __launch_bounds__(256) void zero_kernel(float4* __restrict__ p, int n4) {
    const int i = blockIdx.x * blockDim.x + threadIdx.x;
    if (i < n4) p[i] = float4{0.f, 0.f, 0.f, 0.f};
}

// m = (h[src] + emb_rel[etype]) @ Wn ; atomically scatter-add into agg[dst]
__global__ __launch_bounds__(256)
void edge_kernel(const float* __restrict__ h, const float* __restrict__ emb,
                 const float* __restrict__ Wn, const int* __restrict__ src,
                 const int* __restrict__ dst, const int* __restrict__ etype,
                 float* __restrict__ agg) {
    const int lane = threadIdx.x & 31;
    const int r = lane & 15, hh = lane >> 4;
    const int wave   = blockIdx.x * (blockDim.x >> 5) + (threadIdx.x >> 5);
    const int nwaves = gridDim.x * (blockDim.x >> 5);

    Frag B[4][2];
    load_wfrags(Wn, lane, B);

    const int ntiles = NE / 16;
    for (int tile = wave; tile < ntiles; tile += nwaves) {
        const unsigned e  = (unsigned)tile * 16u + (unsigned)r;
        const unsigned s  = (unsigned)src[e];
        const unsigned et = (unsigned)etype[e];
        const float2* hp = (const float2*)(h + s * (unsigned)DD);
        const float2* rp = (const float2*)(emb + et * (unsigned)DD);

        Frag A[2];
#pragma unroll
        for (int f = 0; f < 2; ++f)
#pragma unroll
            for (int v = 0; v < 8; ++v) {
                const int k = frag_k(f, v, hh);
                const float2 x = hp[k >> 1];
                const float2 y = rp[k >> 1];
                A[f].p[v] = pk_bf16(x.x + y.x, x.y + y.y);
            }

        v8f C[4];
#pragma unroll
        for (int t = 0; t < 4; ++t) {
            v8f c = {0.f, 0.f, 0.f, 0.f, 0.f, 0.f, 0.f, 0.f};
            c = wmma_bf(A[0], B[t][0], c);
            c = wmma_bf(A[1], B[t][1], c);
            C[t] = c;
        }

        unsigned drow[8];
#pragma unroll
        for (int m = 0; m < 8; ++m)
            drow[m] = (unsigned)dst[(unsigned)tile * 16u + (unsigned)(hh * 8 + m)];

#pragma unroll
        for (int t = 0; t < 4; ++t)
#pragma unroll
            for (int m = 0; m < 8; ++m) {
                const unsigned off = drow[m] * (unsigned)DD + (unsigned)(t * 16 + r);
                unsafeAtomicAdd(agg + off, C[t][m]);
            }
    }
}

// out = out * norm + h @ loop_weight   (out currently holds agg)
__global__ __launch_bounds__(256)
void node_hn_kernel(const float* __restrict__ h, const float* __restrict__ Wl,
                    const float* __restrict__ norm, float* __restrict__ out) {
    const int lane = threadIdx.x & 31;
    const int r = lane & 15, hh = lane >> 4;
    const int wave   = blockIdx.x * (blockDim.x >> 5) + (threadIdx.x >> 5);
    const int nwaves = gridDim.x * (blockDim.x >> 5);

    Frag B[4][2];
    load_wfrags(Wl, lane, B);

    const int ntiles = NN / 16;
    for (int tile = wave; tile < ntiles; tile += nwaves) {
        const float2* hp =
            (const float2*)(h + ((unsigned)tile * 16u + (unsigned)r) * (unsigned)DD);
        Frag A[2];
        load_afrags(hp, hh, A);

        v8f C[4];
#pragma unroll
        for (int t = 0; t < 4; ++t) {
            v8f c = {0.f, 0.f, 0.f, 0.f, 0.f, 0.f, 0.f, 0.f};
            c = wmma_bf(A[0], B[t][0], c);
            c = wmma_bf(A[1], B[t][1], c);
            C[t] = c;
        }
#pragma unroll
        for (int t = 0; t < 4; ++t)
#pragma unroll
            for (int m = 0; m < 8; ++m) {
                const unsigned nd  = (unsigned)tile * 16u + (unsigned)(hh * 8 + m);
                const unsigned idx = nd * (unsigned)DD + (unsigned)(t * 16 + r);
                out[idx] = out[idx] * norm[nd] + C[t][m];
            }
    }
}

// sig = sigmoid(prev_h @ Ws + bias); out = relu(sig*out + (1-sig)*prev_h)
__global__ __launch_bounds__(256)
void node_final_kernel(const float* __restrict__ prev_h, const float* __restrict__ Ws,
                       const float* __restrict__ bias, float* __restrict__ out) {
    const int lane = threadIdx.x & 31;
    const int r = lane & 15, hh = lane >> 4;
    const int wave   = blockIdx.x * (blockDim.x >> 5) + (threadIdx.x >> 5);
    const int nwaves = gridDim.x * (blockDim.x >> 5);

    Frag B[4][2];
    load_wfrags(Ws, lane, B);

    float bcol[4];
#pragma unroll
    for (int t = 0; t < 4; ++t) bcol[t] = bias[t * 16 + r];

    const int ntiles = NN / 16;
    for (int tile = wave; tile < ntiles; tile += nwaves) {
        const float2* pp =
            (const float2*)(prev_h + ((unsigned)tile * 16u + (unsigned)r) * (unsigned)DD);
        Frag A[2];
        load_afrags(pp, hh, A);

        v8f C[4];
#pragma unroll
        for (int t = 0; t < 4; ++t) {
            v8f c = {0.f, 0.f, 0.f, 0.f, 0.f, 0.f, 0.f, 0.f};
            c = wmma_bf(A[0], B[t][0], c);
            c = wmma_bf(A[1], B[t][1], c);
            C[t] = c;
        }
#pragma unroll
        for (int t = 0; t < 4; ++t)
#pragma unroll
            for (int m = 0; m < 8; ++m) {
                const unsigned nd  = (unsigned)tile * 16u + (unsigned)(hh * 8 + m);
                const unsigned idx = nd * (unsigned)DD + (unsigned)(t * 16 + r);
                const float sig = 1.0f / (1.0f + __expf(-(C[t][m] + bcol[t])));
                const float res = sig * out[idx] + (1.0f - sig) * prev_h[idx];
                out[idx] = res > 0.0f ? res : 0.0f;
            }
    }
}

extern "C" void kernel_launch(void* const* d_in, const int* in_sizes, int n_in,
                              void* d_out, int out_size, void* d_ws, size_t ws_size,
                              hipStream_t stream) {
    const float* h      = (const float*)d_in[0];
    const float* norm   = (const float*)d_in[1];
    const float* prev_h = (const float*)d_in[2];
    const float* emb    = (const float*)d_in[3];
    const float* Wn     = (const float*)d_in[4];
    const float* Wl     = (const float*)d_in[5];
    const float* Ws     = (const float*)d_in[6];
    const float* bias   = (const float*)d_in[7];
    const int*   src    = (const int*)d_in[8];
    const int*   dst    = (const int*)d_in[9];
    const int*   etype  = (const int*)d_in[10];
    float* out = (float*)d_out;

    const int n4 = (NN * DD) / 4;
    zero_kernel<<<(n4 + 255) / 256, 256, 0, stream>>>((float4*)out, n4);
    edge_kernel<<<512, 256, 0, stream>>>(h, emb, Wn, src, dst, etype, out);
    node_hn_kernel<<<256, 256, 0, stream>>>(h, Wl, norm, out);
    node_final_kernel<<<256, 256, 0, stream>>>(prev_h, Ws, bias, out);
}
